// GATV2_62345745269321
// MI455X (gfx1250) — matrix operand order repaired and checked by reading, not compiled, not run on hardware
//
#include <hip/hip_runtime.h>
#include <hip/hip_bf16.h>
#include <stdint.h>

#define N_NODES   100000
#define N_EDGES   1600000
#define E_TOT     (N_EDGES + N_NODES)   // self loops appended, PyG style
#define N_GRAPHS  64
#define D_IN      200
#define HEADS     5
#define CH        32
#define HID       160
#define K1PAD     224                   // D_IN padded to multiple of 32
#define NT        10                    // HID / 16 column tiles
#define MT        (N_NODES / 16)        // 6250 row tiles
#define NEG_SLOPE 0.2f

typedef __attribute__((ext_vector_type(16))) __bf16 v16bf;
typedef __attribute__((ext_vector_type(8)))  float  v8f;
typedef int v4i_vec __attribute__((vector_size(16)));   // matches builtin param type

union AFrag { v16bf v; unsigned u[8]; };
union BFrag { v16bf v; uint4 q[2]; };

#if defined(__gfx1250__) && __has_builtin(__builtin_amdgcn_global_load_async_to_lds_b128)
#define HAVE_ASYNC_LDS 1
#else
#define HAVE_ASYNC_LDS 0
#endif

// ---- monotonic float<->uint encoding so atomicMax(u32) == float max ----
__device__ __forceinline__ unsigned f2mono(float f) {
    unsigned u = __float_as_uint(f);
    return (u & 0x80000000u) ? ~u : (u | 0x80000000u);
}
__device__ __forceinline__ float mono2f(unsigned m) {
    return __uint_as_float((m & 0x80000000u) ? (m & 0x7FFFFFFFu) : ~m);
}

// ---------------- layer-1 input: f32 [N,200] -> bf16 [N,224] (zero pad) ----
__global__ void cvt_x_kernel(const float* __restrict__ x, __bf16* __restrict__ A) {
    long long i = (long long)blockIdx.x * blockDim.x + threadIdx.x;
    if (i >= (long long)N_NODES * K1PAD) return;
    int col = (int)(i % K1PAD);
    long long row = i / K1PAD;
    float v = (col < D_IN) ? x[row * D_IN + col] : 0.0f;
    A[i] = (__bf16)v;
}

// ---------------- pack W (f32 [K,160]) into per-lane WMMA B-frag layout ----
// B tile 32x16 bf16: lane = half*16+n, VGPR v holds K = kt*32 + half*16 + 2v (+1)
__global__ void pack_w_kernel(const float* __restrict__ W, int K,
                              unsigned* __restrict__ Bp, int ktiles) {
    int idx = blockIdx.x * blockDim.x + threadIdx.x;
    if (idx >= ktiles * NT * 256) return;
    int j    = idx & 7;
    int lane = (idx >> 3) & 31;
    int nt   = (idx >> 8) % NT;
    int kt   = idx / (NT * 256);
    int n = lane & 15, half = lane >> 4;
    int k0  = kt * 32 + half * 16 + 2 * j;
    int col = nt * 16 + n;
    float w0 = (k0     < K) ? W[(long long)k0 * HID + col]       : 0.0f;
    float w1 = (k0 + 1 < K) ? W[(long long)(k0 + 1) * HID + col] : 0.0f;
    union { __bf16 h[2]; unsigned u; } p;
    p.h[0] = (__bf16)w0; p.h[1] = (__bf16)w1;
    Bp[idx] = p.u;
}

// ---------------- WMMA GEMM: C[N_NODES,160] = A[N_NODES,K] * W[K,160] ------
// block = one 16-row A panel (async-staged into LDS), 10 waves = 10 N-tiles
__global__ __launch_bounds__(320) void gemm_wmma_kernel(
    const __bf16* __restrict__ A, const unsigned* __restrict__ Bp,
    float* __restrict__ C, int ldaDw, int ktiles) {
    __shared__ unsigned panel[16 * (K1PAD / 2)];   // up to 16 x 224 bf16 = 7KB
    const int mtile = blockIdx.x;
    const int tid = threadIdx.x;
    const int ndw = 16 * ldaDw;

#if HAVE_ASYNC_LDS
    // gfx1250 async global->LDS copy: no VGPR round trip, tracked by ASYNCcnt
    {
        const char* Ag = (const char*)A + (size_t)mtile * 16 * ldaDw * 4;
        char* Ls = (char*)panel;
        for (int i = tid; i < (ndw >> 2); i += 320) {
            __builtin_amdgcn_global_load_async_to_lds_b128(
                (__attribute__((address_space(1))) v4i_vec*)(Ag + (size_t)i * 16),
                (__attribute__((address_space(3))) v4i_vec*)(Ls + (size_t)i * 16),
                0, 0);
        }
    }
    asm volatile("s_wait_asynccnt 0x0" ::: "memory");
#else
    {
        const unsigned* Ad = (const unsigned*)A;
        const long long abase = (long long)mtile * 16 * ldaDw;
        for (int i = tid; i < ndw; i += 320) panel[i] = Ad[abase + i];
    }
#endif
    __syncthreads();

    const int lane = tid & 31;
    const int wave = tid >> 5;              // = ntile (0..9)
    const int m = lane & 15, half = lane >> 4;
    v8f acc = {};
    for (int kt = 0; kt < ktiles; ++kt) {
        AFrag a; BFrag b;
#pragma unroll
        for (int v = 0; v < 8; ++v) {
            int dwoff = v + 8 * half;   // contiguous K layout: K = half*16 + 2v (+1), matches packed B
            a.u[v] = panel[m * ldaDw + kt * 16 + dwoff];
        }
        const uint4* bq = (const uint4*)(Bp + (((long long)kt * NT + wave) * 32 + lane) * 8);
        b.q[0] = bq[0]; b.q[1] = bq[1];
        acc = __builtin_amdgcn_wmma_f32_16x16x32_bf16(
            false, a.v, false, b.v, (short)0, acc, false, false);
    }
    const int col = wave * 16 + m;
    const long long rb = (long long)mtile * 16 + half * 8;
#pragma unroll
    for (int r = 0; r < 8; ++r)
        C[(rb + r) * HID + col] = acc[r];
}

// ------- edge pass 1: wave-per-edge, lane-per-channel, coalesced gathers ----
__global__ void edge_score_kernel(const long long* __restrict__ ei,
                                  const float* __restrict__ xl,
                                  const float* __restrict__ xr,
                                  const float* __restrict__ att,
                                  float* __restrict__ score,
                                  unsigned* __restrict__ smax) {
    long long e = (long long)blockIdx.x * 8 + (threadIdx.x >> 5);
    if (e >= E_TOT) return;
    int lane = threadIdx.x & 31;
    long long s, d;
    if (e < N_EDGES) { s = ei[e]; d = ei[N_EDGES + e]; }
    else             { s = e - N_EDGES; d = s; }
    const float* pl = xl + s * HID;
    const float* pr = xr + d * HID;
#pragma unroll
    for (int hh = 0; hh < HEADS; ++hh) {
        int c = hh * CH + lane;
        float v = pl[c] + pr[c];                  // one coalesced 128B line each
        v = v > 0.f ? v : NEG_SLOPE * v;
        float contrib = att[c] * v;
#pragma unroll
        for (int off = 16; off > 0; off >>= 1)    // wave32 xor-reduction
            contrib += __shfl_xor(contrib, off, 32);
        if (lane == 0) {
            score[e * HEADS + hh] = contrib;
            atomicMax(&smax[d * HEADS + hh], f2mono(contrib));
        }
    }
}

// ---------------- edge pass 2: exp + segment sum (no gathers) --------------
__global__ void edge_exp_kernel(const long long* __restrict__ ei,
                                float* __restrict__ score,
                                const unsigned* __restrict__ smax,
                                float* __restrict__ denom) {
    long long idx = (long long)blockIdx.x * blockDim.x + threadIdx.x;
    if (idx >= (long long)E_TOT * HEADS) return;
    int hh = (int)(idx % HEADS);
    long long e = idx / HEADS;
    long long d = (e < N_EDGES) ? ei[N_EDGES + e] : (e - N_EDGES);
    float ex = __expf(score[idx] - mono2f(smax[d * HEADS + hh]));
    score[idx] = ex;
    atomicAdd(&denom[d * HEADS + hh], ex);
}

// ------- edge pass 3: wave-per-edge weighted scatter-add (coalesced) -------
__global__ void edge_agg_kernel(const long long* __restrict__ ei,
                                const float* __restrict__ score,
                                const float* __restrict__ denom,
                                const float* __restrict__ xl,
                                float* __restrict__ hout) {
    long long e = (long long)blockIdx.x * 8 + (threadIdx.x >> 5);
    if (e >= E_TOT) return;
    int lane = threadIdx.x & 31;
    long long s, d;
    if (e < N_EDGES) { s = ei[e]; d = ei[N_EDGES + e]; }
    else             { s = e - N_EDGES; d = s; }
    const float* pl = xl + s * HID;
    float* po = hout + d * HID;
#pragma unroll
    for (int hh = 0; hh < HEADS; ++hh) {
        float alpha = score[e * HEADS + hh] / (denom[d * HEADS + hh] + 1e-16f);
        int c = hh * CH + lane;
        atomicAdd(&po[c], alpha * pl[c]);         // 32 consecutive addresses
    }
}

// ---------------- bias add; emit bf16 A matrix for next layer -------------
__global__ void bias_kernel(float* __restrict__ h, const float* __restrict__ b,
                            __bf16* __restrict__ Anext, int writeNext) {
    long long i = (long long)blockIdx.x * blockDim.x + threadIdx.x;
    if (i >= (long long)N_NODES * HID) return;
    float v = h[i] + b[(int)(i % HID)];
    h[i] = v;
    if (writeNext) Anext[i] = (__bf16)v;
}

// ---------------- global mean pool (sums + counts) ------------------------
__global__ void pool_kernel(const float* __restrict__ h,
                            const long long* __restrict__ batch,
                            float* __restrict__ pooled, float* __restrict__ counts) {
    long long i = (long long)blockIdx.x * blockDim.x + threadIdx.x;
    if (i >= (long long)N_NODES * HID) return;
    int col = (int)(i % HID);
    long long node = i / HID;
    int g = (int)batch[node];
    atomicAdd(&pooled[g * HID + col], h[i]);
    if (col == 0) atomicAdd(&counts[g], 1.0f);
}

// ---------------- final linear + log_softmax ------------------------------
__global__ void logits_kernel(const float* __restrict__ pooled,
                              const float* __restrict__ counts,
                              const float* __restrict__ Wlin,
                              const float* __restrict__ blin,
                              float* __restrict__ out) {
    int g = threadIdx.x;
    if (g >= N_GRAPHS) return;
    float cnt = counts[g]; cnt = cnt > 1.0f ? cnt : 1.0f;
    float z0 = blin[0], z1 = blin[1];
    for (int k = 0; k < HID; ++k) {
        float p = pooled[g * HID + k] / cnt;
        z0 += p * Wlin[k * 2 + 0];
        z1 += p * Wlin[k * 2 + 1];
    }
    float m = fmaxf(z0, z1);
    float lse = m + __logf(__expf(z0 - m) + __expf(z1 - m));
    out[g * 2 + 0] = z0 - lse;
    out[g * 2 + 1] = z1 - lse;
}

extern "C" void kernel_launch(void* const* d_in, const int* in_sizes, int n_in,
                              void* d_out, int out_size, void* d_ws, size_t ws_size,
                              hipStream_t stream) {
    const float*     x     = (const float*)d_in[0];
    const long long* eidx  = (const long long*)d_in[1];   // int64 per reference
    const long long* batch = (const long long*)d_in[3];   // int64
    const float* Wl[3]  = {(const float*)d_in[4],  (const float*)d_in[8],  (const float*)d_in[12]};
    const float* Wr[3]  = {(const float*)d_in[5],  (const float*)d_in[9],  (const float*)d_in[13]};
    const float* att[3] = {(const float*)d_in[6],  (const float*)d_in[10], (const float*)d_in[14]};
    const float* bb[3]  = {(const float*)d_in[7],  (const float*)d_in[11], (const float*)d_in[15]};
    const float* Wlin   = (const float*)d_in[16];
    const float* blin   = (const float*)d_in[17];
    float* out = (float*)d_out;

    // ----- workspace carve-out -----
    char* base = (char*)d_ws;
    size_t off = 0;
    auto carve = [&](size_t bytes) -> char* {
        char* p = base + off;
        off = (off + bytes + 255) & ~(size_t)255;
        return p;
    };
    __bf16*   Abf    = (__bf16*)  carve((size_t)N_NODES * K1PAD * 2);
    float*    xl     = (float*)   carve((size_t)N_NODES * HID * 4);
    float*    xr     = (float*)   carve((size_t)N_NODES * HID * 4);
    float*    hout   = (float*)   carve((size_t)N_NODES * HID * 4);
    float*    score  = (float*)   carve((size_t)E_TOT * HEADS * 4);
    unsigned* smax   = (unsigned*)carve((size_t)N_NODES * HEADS * 4);
    float*    denom  = (float*)   carve((size_t)N_NODES * HEADS * 4);
    unsigned* Bp0    = (unsigned*)carve(7 * NT * 256 * 4);
    unsigned* Bp1    = (unsigned*)carve(7 * NT * 256 * 4);
    float*    pooled = (float*)   carve((size_t)N_GRAPHS * HID * 4);
    float*    counts = (float*)   carve((size_t)N_GRAPHS * 4);

    const int K[3]      = {D_IN, HID, HID};
    const int ktiles[3] = {K1PAD / 32, HID / 32, HID / 32};   // 7, 5, 5
    const int ldaDw[3]  = {K1PAD / 2, HID / 2, HID / 2};      // 112, 80, 80

    // layer-1 input conversion (padded bf16)
    {
        long long n = (long long)N_NODES * K1PAD;
        cvt_x_kernel<<<(unsigned)((n + 255) / 256), 256, 0, stream>>>(x, Abf);
    }

    const unsigned edgeWaveBlocks = (unsigned)((E_TOT + 7) / 8);          // 8 waves/block
    const long long nEdgeThreads  = (long long)E_TOT * HEADS;
    const unsigned edgeThrBlocks  = (unsigned)((nEdgeThreads + 255) / 256);
    const long long nNodeElems    = (long long)N_NODES * HID;
    const unsigned nodeBlocks     = (unsigned)((nNodeElems + 255) / 256);

    for (int L = 0; L < 3; ++L) {
        // pack weights into WMMA B layout
        int packN = ktiles[L] * NT * 256;
        pack_w_kernel<<<(packN + 255) / 256, 256, 0, stream>>>(Wl[L], K[L], Bp0, ktiles[L]);
        pack_w_kernel<<<(packN + 255) / 256, 256, 0, stream>>>(Wr[L], K[L], Bp1, ktiles[L]);

        // dense transforms via WMMA
        gemm_wmma_kernel<<<MT, 320, 0, stream>>>(Abf, Bp0, xl, ldaDw[L], ktiles[L]);
        gemm_wmma_kernel<<<MT, 320, 0, stream>>>(Abf, Bp1, xr, ldaDw[L], ktiles[L]);

        // clear segment buffers + output accumulator
        (void)hipMemsetAsync(smax,  0, (size_t)N_NODES * HEADS * 4, stream);
        (void)hipMemsetAsync(denom, 0, (size_t)N_NODES * HEADS * 4, stream);
        (void)hipMemsetAsync(hout,  0, (size_t)N_NODES * HID * 4, stream);

        // attention + segment softmax + aggregation
        edge_score_kernel<<<edgeWaveBlocks, 256, 0, stream>>>(eidx, xl, xr, att[L], score, smax);
        edge_exp_kernel  <<<edgeThrBlocks,  256, 0, stream>>>(eidx, score, smax, denom);
        edge_agg_kernel  <<<edgeWaveBlocks, 256, 0, stream>>>(eidx, score, denom, xl, hout);

        // bias; prepare bf16 input for next layer (lda = 160)
        bias_kernel<<<nodeBlocks, 256, 0, stream>>>(hout, bb[L], Abf, L < 2 ? 1 : 0);
    }

    // mean pool + classifier head
    (void)hipMemsetAsync(pooled, 0, (size_t)N_GRAPHS * HID * 4, stream);
    (void)hipMemsetAsync(counts, 0, (size_t)N_GRAPHS * 4, stream);
    pool_kernel<<<nodeBlocks, 256, 0, stream>>>(hout, batch, pooled, counts);
    logits_kernel<<<1, 64, 0, stream>>>(pooled, counts, Wlin, blin, out);
}
